// MHSAwithQS_53291954209339
// MI455X (gfx1250) — matrix operand order repaired and checked by reading, not compiled, not run on hardware
//
#include <hip/hip_runtime.h>
#include <hip/hip_bf16.h>
#include <stdint.h>

typedef __bf16 bf16_t;
typedef __attribute__((ext_vector_type(16))) __bf16 v16bf;
typedef __attribute__((ext_vector_type(8)))  float  v8f;

#define B_  16
#define S_  1024
#define E_  512
#define H_  8
#define D_  64

// ---------------------------------------------------------------------------
// WMMA helpers (CDNA5 wave32, 16x16x32 bf16 -> f32 accumulate)
// ---------------------------------------------------------------------------
__device__ __forceinline__ v8f wmma_bf16(v16bf a, v16bf b, v8f c) {
  // (neg_a, A, neg_b, B, c_mod, C, reuse_a, reuse_b)
  return __builtin_amdgcn_wmma_f32_16x16x32_bf16(false, a, false, b, (short)0, c,
                                                 false, false);
}

union Pack16 { uint4 u[2]; v16bf v; __bf16 h[16]; };
union Pack8  { uint4 u;    __bf16 h[8]; };

// A-operand (16x32 bf16, M rows x K): contiguous k order, same as load_b_frag
//   lanes 0-15 : row = row0+lane,    K = k0+[0..15] contiguous
//   lanes 16-31: row = row0+lane-16, K = k0+[16..31] contiguous
__device__ __forceinline__ v16bf load_a_frag(const bf16_t* base, int row0, int ld,
                                             int k0, int lane) {
  int r  = row0 + (lane & 15);
  int kk = k0 + ((lane >> 4) << 4);
  const bf16_t* p = base + (size_t)r * ld + kk;
  Pack16 u;
  u.u[0] = *(const uint4*)(p);
  u.u[1] = *(const uint4*)(p + 8);
  return u.v;
}

// B-operand (32x16 bf16, K x N cols). B[k][n] = M2[n][k] for row-major M2:
//   lanes 0-15 : col = col0+lane,    K = k0+[0..15] contiguous
//   lanes 16-31: col = col0+lane-16, K = k0+[16..31] contiguous
__device__ __forceinline__ v16bf load_b_frag(const bf16_t* base, int col0, int ld,
                                             int k0, int lane) {
  int n  = col0 + (lane & 15);
  int kk = k0 + ((lane >> 4) << 4);
  const bf16_t* p = base + (size_t)n * ld + kk;
  Pack16 u;
  u.u[0] = *(const uint4*)(p);
  u.u[1] = *(const uint4*)(p + 8);
  return u.v;
}

// ---------------------------------------------------------------------------
// K0: fp32 -> bf16 conversion (grid stride)
// ---------------------------------------------------------------------------
__global__ __launch_bounds__(256) void cvt_f32_bf16(const float* __restrict__ src,
                                                    bf16_t* __restrict__ dst, int n) {
  int i = blockIdx.x * blockDim.x + threadIdx.x;
  int stride = gridDim.x * blockDim.x;
  for (; i < n; i += stride) dst[i] = (__bf16)src[i];
}

// K0b: qscale[d] = log2(e) * softplus(qs[d]) * (1/sqrt(D))
__global__ void qscale_kernel(const float* __restrict__ qs, float* __restrict__ out) {
  int i = threadIdx.x;
  if (i < D_) {
    float x  = qs[i];
    float sp = (x > 20.f) ? x : log1pf(__expf(x));
    out[i] = 1.4426950408889634f * sp * 0.125f;   // fold 1/sqrt(64)
  }
}

// ---------------------------------------------------------------------------
// K1: QKV projection.  C^T tiles: A = Wqkv rows [1536,512], B = X rows [16384,512]
// Each wave: 16 samples x 64 outputs, K=512.
// Ping-pong software pipeline (unroll-by-2, two independent fragment sets, no
// register rotation copies): loads for one set issue while the other set's
// 4 WMMAs execute.
// Stores Q (scaled), K, V as bf16 in [B,H,S,D].
// ---------------------------------------------------------------------------
__global__ __launch_bounds__(256) void qkv_gemm(const bf16_t* __restrict__ xb,
                                                const bf16_t* __restrict__ wb,
                                                const float* __restrict__ bias,
                                                const float* __restrict__ qscale,
                                                bf16_t* __restrict__ qws,
                                                bf16_t* __restrict__ kws,
                                                bf16_t* __restrict__ vws) {
  int lane = threadIdx.x & 31;
  int w    = threadIdx.x >> 5;
  int m0 = blockIdx.y * 128 + w * 16;   // sample tile (WMMA columns)
  int n0 = blockIdx.x * 64;             // output tile (WMMA rows)
  v8f acc[4] = {};

  v16bf bf0 = load_b_frag(xb, m0, 512, 0, lane);
  v16bf af0[4];
#pragma unroll
  for (int t = 0; t < 4; ++t) af0[t] = load_a_frag(wb, n0 + t * 16, 512, 0, lane);

  for (int kk = 0; kk < 512; kk += 64) {
    // issue set1 loads (k = kk+32), then consume set0
    __builtin_prefetch(xb + (size_t)(m0 + (lane & 15)) * 512 + kk + 32, 0, 1);
    v16bf bf1 = load_b_frag(xb, m0, 512, kk + 32, lane);
    v16bf af1[4];
#pragma unroll
    for (int t = 0; t < 4; ++t) af1[t] = load_a_frag(wb, n0 + t * 16, 512, kk + 32, lane);
#pragma unroll
    for (int t = 0; t < 4; ++t) acc[t] = wmma_bf16(af0[t], bf0, acc[t]);

    // issue set0 loads (k = kk+64; wrap to 0 on last iter), then consume set1
    int kn = (kk + 64 < 512) ? (kk + 64) : 0;
    bf0 = load_b_frag(xb, m0, 512, kn, lane);
#pragma unroll
    for (int t = 0; t < 4; ++t) af0[t] = load_a_frag(wb, n0 + t * 16, 512, kn, lane);
#pragma unroll
    for (int t = 0; t < 4; ++t) acc[t] = wmma_bf16(af1[t], bf1, acc[t]);
  }

  int m = m0 + (lane & 15);
  int b = m >> 10, s = m & 1023;
#pragma unroll
  for (int t = 0; t < 4; ++t) {
    int nb   = n0 + t * 16 + ((lane >> 4) << 3);  // 8 consecutive outputs per lane
    int part = nb >> 9;                           // 0=Q 1=K 2=V
    int e    = nb & 511;
    int h = e >> 6, d = e & 63;
    Pack8 pk;
#pragma unroll
    for (int r = 0; r < 8; ++r) {
      float v = acc[t][r] + bias[nb + r];
      if (part == 0) v *= qscale[d + r];
      pk.h[r] = (__bf16)v;
    }
    bf16_t* base = (part == 0) ? qws : ((part == 1) ? kws : vws);
    bf16_t* dst  = base + (((size_t)(b * H_ + h) * S_ + s) * D_ + d);
    *(uint4*)dst = pk.u;
  }
}

// ---------------------------------------------------------------------------
// K2: flash attention per (b,h). Computes S^T = K*Q^T and O^T = V^T*P^T so the
// softmax statistics are per-lane (query = lane%16) and the flash rescale is a
// per-lane scalar multiply of the O^T accumulators.
// Block = 8 waves, each wave owns 16 queries. K/V key-chunks (32 keys) are
// cooperatively staged into double-buffered LDS: chunk j+1 is staged while
// chunk j is computed, with one barrier per iteration.
// ---------------------------------------------------------------------------
__global__ __launch_bounds__(256) void attn_kernel(const bf16_t* __restrict__ qws,
                                                   const bf16_t* __restrict__ kws,
                                                   const bf16_t* __restrict__ vws,
                                                   const unsigned char* __restrict__ mask,
                                                   bf16_t* __restrict__ ows) {
  __shared__ bf16_t Kc[2][32 * 64];   // K chunk, row-major [key][dim]
  __shared__ bf16_t Vt[2][64 * 32];   // V chunk transposed [dim][key]
  int lane = threadIdx.x & 31;
  int w    = threadIdx.x >> 5;
  int bh = blockIdx.y;
  int b = bh >> 3, h = bh & 7;
  const bf16_t* Qbh = qws + (size_t)bh * S_ * D_;
  const bf16_t* Kbh = kws + (size_t)bh * S_ * D_;
  const bf16_t* Vbh = vws + (size_t)bh * S_ * D_;
  const unsigned char* mb = mask + (size_t)b * S_;
  int q0 = blockIdx.x * 128 + w * 16;

  // Q^T as B-operand: contiguous loads from Q rows, resident for whole loop
  v16bf qb0 = load_b_frag(Qbh, q0, 64, 0, lane);
  v16bf qb1 = load_b_frag(Qbh, q0, 64, 32, lane);

  v8f o0 = {}, o1 = {}, o2 = {}, o3 = {};
  float mrun = -3.0e38f, lrun = 0.0f;

  int skr = threadIdx.x >> 3, skc = (threadIdx.x & 7) * 8;     // K staging
  int svk = threadIdx.x & 31, svd = (threadIdx.x >> 5) * 8;    // V staging

  // stage chunk 0 into buffer 0
  {
    *(uint4*)&Kc[0][skr * 64 + skc] = *(const uint4*)&Kbh[(size_t)skr * 64 + skc];
    Pack8 vv; vv.u = *(const uint4*)&Vbh[(size_t)svk * 64 + svd];
#pragma unroll
    for (int i = 0; i < 8; ++i) Vt[0][(svd + i) * 32 + svk] = vv.h[i];
  }
  __syncthreads();

  int cur = 0;
  for (int j0 = 0; j0 < S_; j0 += 32, cur ^= 1) {
    // ---- stage next chunk into the other buffer (overlaps compute) ----
    int jn = j0 + 32;
    if (jn < S_) {
      *(uint4*)&Kc[cur ^ 1][skr * 64 + skc] =
          *(const uint4*)&Kbh[(size_t)(jn + skr) * 64 + skc];
      Pack8 vv; vv.u = *(const uint4*)&Vbh[(size_t)(jn + svk) * 64 + svd];
#pragma unroll
      for (int i = 0; i < 8; ++i) Vt[cur ^ 1][(svd + i) * 32 + svk] = vv.h[i];
    }
    const bf16_t* kc = Kc[cur];
    const bf16_t* vt = Vt[cur];

    // key-padding mask bytes for this lane's 8 keys per tile
    int moff = j0 + ((lane >> 4) << 3);
    unsigned long long mm0 = *(const unsigned long long*)(mb + moff);
    unsigned long long mm1 = *(const unsigned long long*)(mb + moff + 16);

    // ---- S^T tiles: [16 keys x 16 queries] each, K-dim = 64 (2 steps) ----
    v8f st0 = {}, st1 = {};
    st0 = wmma_bf16(load_a_frag(kc, 0, 64, 0, lane),  qb0, st0);
    st0 = wmma_bf16(load_a_frag(kc, 0, 64, 32, lane), qb1, st0);
    st1 = wmma_bf16(load_a_frag(kc, 16, 64, 0, lane),  qb0, st1);
    st1 = wmma_bf16(load_a_frag(kc, 16, 64, 32, lane), qb1, st1);

    float s0[8], s1[8];
    float mloc = -3.0e38f;
#pragma unroll
    for (int r = 0; r < 8; ++r) {
      s0[r] = ((mm0 >> (8 * r)) & 1ull) ? -3.0e38f : st0[r];
      s1[r] = ((mm1 >> (8 * r)) & 1ull) ? -3.0e38f : st1[r];
      mloc = fmaxf(mloc, fmaxf(s0[r], s1[r]));
    }
    mloc = fmaxf(mloc, __shfl_xor(mloc, 16, 32));  // other half of query's keys
    float mnew = fmaxf(mrun, mloc);
    float p0[8], p1[8], lsum = 0.f;
#pragma unroll
    for (int r = 0; r < 8; ++r) {
      p0[r] = ((mm0 >> (8 * r)) & 1ull) ? 0.f : __expf(s0[r] - mnew);
      p1[r] = ((mm1 >> (8 * r)) & 1ull) ? 0.f : __expf(s1[r] - mnew);
      lsum += p0[r] + p1[r];
    }
    lsum += __shfl_xor(lsum, 16, 32);
    float c = __expf(mrun - mnew);
    lrun = lrun * c + lsum;
    mrun = mnew;
    o0 *= c; o1 *= c; o2 *= c; o3 *= c;   // per-lane flash rescale

    // ---- build P^T B-fragment via half-wave exchange ----
    float x0[8], x1[8];
#pragma unroll
    for (int r = 0; r < 8; ++r) {
      x0[r] = __shfl_xor(p0[r], 16, 32);
      x1[r] = __shfl_xor(p1[r], 16, 32);
    }
    bool hi = lane >= 16;
    Pack16 pf;
#pragma unroll
    for (int r = 0; r < 8; ++r) {
      pf.h[r]     = (__bf16)(hi ? x1[r] : p0[r]);  // keys j0+16..23 / j0+0..7
      pf.h[8 + r] = (__bf16)(hi ? p1[r] : x0[r]);  // keys j0+24..31 / j0+8..15
    }

    // ---- O^T += V^T * P^T  (4 dim-tiles of 16) ----
    o0 = wmma_bf16(load_a_frag(vt, 0, 32, 0, lane),  pf.v, o0);
    o1 = wmma_bf16(load_a_frag(vt, 16, 32, 0, lane), pf.v, o1);
    o2 = wmma_bf16(load_a_frag(vt, 32, 32, 0, lane), pf.v, o2);
    o3 = wmma_bf16(load_a_frag(vt, 48, 32, 0, lane), pf.v, o3);

    // single barrier: next-chunk staging done AND current buffer free for reuse
    __syncthreads();
  }

  // ---- finalize: O^T lane layout packs into contiguous 16B stores ----
  float inv_l = 1.0f / lrun;
  int q = q0 + (lane & 15);
  int dhalf = (lane >> 4) << 3;
  v8f oo[4] = { o0, o1, o2, o3 };
#pragma unroll
  for (int t = 0; t < 4; ++t) {
    Pack8 pk;
#pragma unroll
    for (int r = 0; r < 8; ++r) pk.h[r] = (__bf16)(oo[t][r] * inv_l);
    bf16_t* dst = ows + ((size_t)(b * S_ + q) * E_ + h * D_ + t * 16 + dhalf);
    *(uint4*)dst = pk.u;
  }
}

// ---------------------------------------------------------------------------
// K3: output projection. C^T tiles: A = Wout rows [512,512], B = O rows.
// Same ping-pong pipeline; fp32 + bias straight to d_out.
// ---------------------------------------------------------------------------
__global__ __launch_bounds__(256) void outproj_gemm(const bf16_t* __restrict__ ob,
                                                    const bf16_t* __restrict__ wb,
                                                    const float* __restrict__ bias,
                                                    float* __restrict__ out) {
  int lane = threadIdx.x & 31;
  int w    = threadIdx.x >> 5;
  int m0 = blockIdx.y * 128 + w * 16;
  int n0 = blockIdx.x * 64;
  v8f acc[4] = {};

  v16bf bf0 = load_b_frag(ob, m0, 512, 0, lane);
  v16bf af0[4];
#pragma unroll
  for (int t = 0; t < 4; ++t) af0[t] = load_a_frag(wb, n0 + t * 16, 512, 0, lane);

  for (int kk = 0; kk < 512; kk += 64) {
    __builtin_prefetch(ob + (size_t)(m0 + (lane & 15)) * 512 + kk + 32, 0, 1);
    v16bf bf1 = load_b_frag(ob, m0, 512, kk + 32, lane);
    v16bf af1[4];
#pragma unroll
    for (int t = 0; t < 4; ++t) af1[t] = load_a_frag(wb, n0 + t * 16, 512, kk + 32, lane);
#pragma unroll
    for (int t = 0; t < 4; ++t) acc[t] = wmma_bf16(af0[t], bf0, acc[t]);

    int kn = (kk + 64 < 512) ? (kk + 64) : 0;
    bf0 = load_b_frag(ob, m0, 512, kn, lane);
#pragma unroll
    for (int t = 0; t < 4; ++t) af0[t] = load_a_frag(wb, n0 + t * 16, 512, kn, lane);
#pragma unroll
    for (int t = 0; t < 4; ++t) acc[t] = wmma_bf16(af1[t], bf1, acc[t]);
  }

  int m = m0 + (lane & 15);
#pragma unroll
  for (int t = 0; t < 4; ++t) {
    int nb = n0 + t * 16 + ((lane >> 4) << 3);
    float* dst = out + (size_t)m * 512 + nb;
#pragma unroll
    for (int r = 0; r < 8; ++r) dst[r] = acc[t][r] + bias[nb + r];
  }
}

// ---------------------------------------------------------------------------
extern "C" void kernel_launch(void* const* d_in, const int* in_sizes, int n_in,
                              void* d_out, int out_size, void* d_ws, size_t ws_size,
                              hipStream_t stream) {
  (void)in_sizes; (void)n_in; (void)out_size; (void)ws_size;
  const float* x            = (const float*)d_in[0];
  const unsigned char* mask = (const unsigned char*)d_in[1];
  const float* wqkv         = (const float*)d_in[2];
  const float* bqkv         = (const float*)d_in[3];
  const float* wout         = (const float*)d_in[4];
  const float* bout         = (const float*)d_in[5];
  const float* qs           = (const float*)d_in[6];
  float* out = (float*)d_out;

  char* ws = (char*)d_ws;
  size_t off = 0;
  auto alloc = [&](size_t bytes) {
    char* p = ws + off;
    off += (bytes + 255) & ~(size_t)255;
    return (void*)p;
  };
  bf16_t* xb     = (bf16_t*)alloc((size_t)B_ * S_ * E_ * 2);
  bf16_t* wqb    = (bf16_t*)alloc((size_t)3 * E_ * E_ * 2);
  bf16_t* wob    = (bf16_t*)alloc((size_t)E_ * E_ * 2);
  bf16_t* qws    = (bf16_t*)alloc((size_t)B_ * H_ * S_ * D_ * 2);
  bf16_t* kws    = (bf16_t*)alloc((size_t)B_ * H_ * S_ * D_ * 2);
  bf16_t* vws    = (bf16_t*)alloc((size_t)B_ * S_ * E_ * 2);
  bf16_t* ows    = (bf16_t*)alloc((size_t)B_ * S_ * E_ * 2);
  float*  qscale = (float*)alloc(D_ * sizeof(float));

  cvt_f32_bf16<<<1024, 256, 0, stream>>>(x, xb, B_ * S_ * E_);
  cvt_f32_bf16<<<512, 256, 0, stream>>>(wqkv, wqb, 3 * E_ * E_);
  cvt_f32_bf16<<<256, 256, 0, stream>>>(wout, wob, E_ * E_);
  qscale_kernel<<<1, 64, 0, stream>>>(qs, qscale);

  // QKV: M=16384 samples (128/block) x N=1536 (64/block)
  qkv_gemm<<<dim3(24, 128), 256, 0, stream>>>(xb, wqb, bqkv, qscale, qws, kws, vws);
  // attention: (B*H=128) x (1024/128 query blocks)
  attn_kernel<<<dim3(8, 128), 256, 0, stream>>>(qws, kws, vws, mask, ows);
  // out-proj: M=16384 x N=512
  outproj_gemm<<<dim3(8, 128), 256, 0, stream>>>(ows, wob, bout, out);
}
